// TA_LSTMCell_7344394076758
// MI455X (gfx1250) — matrix operand order repaired and checked by reading, not compiled, not run on hardware
//
#include <hip/hip_runtime.h>
#include <hip/hip_bf16.h>
#include <stdint.h>

typedef __attribute__((ext_vector_type(16))) __bf16 v16bf;
typedef __attribute__((ext_vector_type(8)))  __bf16 v8bf;
typedef __attribute__((ext_vector_type(4)))  __bf16 v4bf;
typedef __attribute__((ext_vector_type(8)))  float  v8f;
typedef __attribute__((ext_vector_type(4)))  int    v4i;

#define BATCH 4096
#define IN_DIM 512
#define HID   1024
#define LDW   72   // LDS row stride in bf16 (64 data + 8 pad, conflict-free)

#if defined(__gfx1250__)
#  if __has_builtin(__builtin_amdgcn_global_load_async_to_lds_b128) && \
      __has_builtin(__builtin_amdgcn_s_wait_asynccnt)
#    define USE_ASYNC 1
#  else
#    define USE_ASYNC 0
#    warning "gfx1250 async-to-LDS builtins not found; using synchronous LDS staging fallback"
#  endif
#else
#  define USE_ASYNC 0
#endif

union FragBF { v16bf v; v8bf h[2]; };

__device__ __forceinline__ v8f wmma_bf16(v8f acc, v16bf a, v16bf b) {
  // (neg_a, A, neg_b, B, c_mod, C, reuse_a, reuse_b)
  return __builtin_amdgcn_wmma_f32_16x16x32_bf16(false, a, false, b, (short)0, acc,
                                                 false, false);
}

// async copy of 16 bytes global -> LDS (ASYNCcnt-tracked); sync fallback otherwise
__device__ __forceinline__ void async_b128(const __bf16* g, __bf16* l) {
#if USE_ASYNC
  auto gp = reinterpret_cast<__attribute__((address_space(1))) v4i*>(
      reinterpret_cast<uintptr_t>(g));
  auto lp = reinterpret_cast<__attribute__((address_space(3))) v4i*>(
      (unsigned int)reinterpret_cast<uintptr_t>(l));
  __builtin_amdgcn_global_load_async_to_lds_b128(gp, lp, 0, 0);
#else
  *(v8bf*)l = *(const v8bf*)g;
#endif
}

template <int N>
__device__ __forceinline__ void wait_async() {
#if USE_ASYNC
  __builtin_amdgcn_s_wait_asynccnt(N);
#endif
}

__device__ __forceinline__ float sigm(float x) {
  return 1.0f / (1.0f + __expf(-x));
}

__device__ __forceinline__ float fast_tanh(float x) {
  float e = __expf(-2.0f * fabsf(x));
  float r = (1.0f - e) / (1.0f + e);
  return copysignf(r, x);
}

// ---------------- prep kernels ----------------

__global__ void cvt_bf16_kernel(const float* __restrict__ s, __bf16* __restrict__ d, int n) {
  int i = (blockIdx.x * blockDim.x + threadIdx.x) * 4;
  if (i < n) {
    float4 f = *(const float4*)(s + i);
    v4bf o;
    o[0] = (__bf16)f.x; o[1] = (__bf16)f.y; o[2] = (__bf16)f.z; o[3] = (__bf16)f.w;
    *(v4bf*)(d + i) = o;
  }
}

// dst[h][k] = src[k][h]  (W_decomp^T, f32 -> bf16)
__global__ void transpose_cvt_kernel(const float* __restrict__ src, __bf16* __restrict__ dst) {
  __shared__ float tile[32][33];
  int tx = threadIdx.x & 31;
  int ty = threadIdx.x >> 5;              // 0..7
  int x0 = blockIdx.x * 32;               // src col
  int y0 = blockIdx.y * 32;               // src row
  #pragma unroll
  for (int r = ty; r < 32; r += 8)
    tile[r][tx] = src[(size_t)(y0 + r) * HID + x0 + tx];
  __syncthreads();
  #pragma unroll
  for (int r = ty; r < 32; r += 8)
    dst[(size_t)(x0 + r) * HID + y0 + tx] = (__bf16)tile[tx][r];
}

__global__ void bias_sum_kernel(const float* __restrict__ a, const float* __restrict__ b,
                                float* __restrict__ o, int n) {
  int i = blockIdx.x * blockDim.x + threadIdx.x;
  if (i < n) o[i] = a[i] + b[i];
}

// ---------------- fused T-LSTM kernel ----------------
// Tile: 64 (batch) x 64 (hidden) per workgroup; 8 waves of 32; wave owns 16x32.
// K-step 64, double-buffered LDS, async global->LDS staging overlapped with WMMA.

__global__ __launch_bounds__(256) void tlstm_fused(
    const __bf16* __restrict__ inbf, const __bf16* __restrict__ hxbf,
    const __bf16* __restrict__ cxbf,
    const float* __restrict__ tvec, const float* __restrict__ cx,
    const __bf16* __restrict__ wih, const __bf16* __restrict__ whh,
    const __bf16* __restrict__ wdt, const float* __restrict__ bias4,
    const float* __restrict__ bdec,
    float* __restrict__ hy, float* __restrict__ cy) {

  __shared__ __align__(16) __bf16 As[2][64][LDW];
  __shared__ __align__(16) __bf16 Bs[2][4][64][LDW];

  const int tid = threadIdx.x;
  const int lane = tid & 31;
  const int wid = tid >> 5;          // 0..7
  const int wm = wid & 3;            // 4 strips of 16 rows
  const int wn = wid >> 2;           // 2 strips of 32 cols
  const int half16 = lane >> 4;      // 0/1
  const int l15 = lane & 15;

  const int row0 = blockIdx.y * 64;  // batch base
  const int col0 = blockIdx.x * 64;  // hidden base

  v8f accg[4][2];
  v8f accd[2];
  v8f zero = {0.f, 0.f, 0.f, 0.f, 0.f, 0.f, 0.f, 0.f};
  #pragma unroll
  for (int g = 0; g < 4; g++) { accg[g][0] = zero; accg[g][1] = zero; }
  accd[0] = zero; accd[1] = zero;

  // -------- staging helpers (pure bf16 copies, async-to-LDS, 64-wide K) --------
  // decomp: 2 A + 2 B = 4 x b128 per thread
  auto stage_dec = [&](int buf, int k0) {
    int tr = tid >> 2, tc = (tid & 3) * 16;
    const __bf16* pa = cxbf + (size_t)(row0 + tr) * HID + k0 + tc;
    async_b128(pa,     &As[buf][tr][tc]);
    async_b128(pa + 8, &As[buf][tr][tc + 8]);
    const __bf16* pb = wdt + (size_t)(col0 + tr) * HID + k0 + tc;
    async_b128(pb,     &Bs[buf][0][tr][tc]);
    async_b128(pb + 8, &Bs[buf][0][tr][tc + 8]);
  };
  // gates: 2 A + 8 B = 10 x b128 per thread; unified K over [input | hx]
  auto stage_gate = [&](int buf, int k0) {
    const __bf16* A; const __bf16* W; int K, kk;
    if (k0 < IN_DIM) { A = inbf; W = wih; K = IN_DIM; kk = k0; }
    else             { A = hxbf; W = whh; K = HID;    kk = k0 - IN_DIM; }
    {
      int tr = tid >> 2, tc = (tid & 3) * 16;
      const __bf16* pa = A + (size_t)(row0 + tr) * K + kk + tc;
      async_b128(pa,     &As[buf][tr][tc]);
      async_b128(pa + 8, &As[buf][tr][tc + 8]);
    }
    {
      int g = tid >> 6, n = tid & 63;
      const __bf16* p = W + (size_t)(g * HID + col0 + n) * K + kk;
      #pragma unroll
      for (int c = 0; c < 8; c++)
        async_b128(p + c * 8, &Bs[buf][g][n][c * 8]);
    }
  };

  // -------- compute helpers (ks = 0 or 32 within the 64-wide tile) --------
  auto load_a = [&](int buf, int ks) {
    FragBF a;
    a.h[0] = *(const v8bf*)&As[buf][wm * 16 + l15][ks + 8 * half16];
    a.h[1] = *(const v8bf*)&As[buf][wm * 16 + l15][ks + 16 + 8 * half16];
    return a;
  };
  auto load_b = [&](int buf, int g, int nt, int ks) {
    FragBF b;
    b.h[0] = *(const v8bf*)&Bs[buf][g][wn * 32 + nt * 16 + l15][ks + 8 * half16];
    b.h[1] = *(const v8bf*)&Bs[buf][g][wn * 32 + nt * 16 + l15][ks + 16 + 8 * half16];
    return b;
  };

  // ---- Pass 1: decomp  C_ST_pre = cx @ W_decomp  (wdt = W_decomp^T, NT form) ----
  stage_dec(0, 0);
  for (int k0 = 0; k0 < HID; k0 += 64) {
    int cur = (k0 >> 6) & 1;
    if (k0 + 64 < HID) { stage_dec(cur ^ 1, k0 + 64); wait_async<4>(); }
    else               { wait_async<0>(); }
    __syncthreads();
    #pragma unroll
    for (int ks = 0; ks < 64; ks += 32) {
      FragBF a = load_a(cur, ks);
      FragBF b0 = load_b(cur, 0, 0, ks);
      FragBF b1 = load_b(cur, 0, 1, ks);
      accd[0] = wmma_bf16(accd[0], a.v, b0.v);
      accd[1] = wmma_bf16(accd[1], a.v, b1.v);
    }
    __syncthreads();
  }

  // ---- Pass 2: gates over unified K = [input(512) | hx(1024)] ----
  stage_gate(0, 0);
  for (int k0 = 0; k0 < IN_DIM + HID; k0 += 64) {
    int cur = (k0 >> 6) & 1;
    if (k0 + 64 < IN_DIM + HID) { stage_gate(cur ^ 1, k0 + 64); wait_async<10>(); }
    else                        { wait_async<0>(); }
    __syncthreads();
    #pragma unroll
    for (int ks = 0; ks < 64; ks += 32) {
      FragBF a = load_a(cur, ks);
      #pragma unroll
      for (int g = 0; g < 4; g++) {
        FragBF b0 = load_b(cur, g, 0, ks);
        FragBF b1 = load_b(cur, g, 1, ks);
        accg[g][0] = wmma_bf16(accg[g][0], a.v, b0.v);
        accg[g][1] = wmma_bf16(accg[g][1], a.v, b1.v);
      }
    }
    __syncthreads();
  }

  // ---- Epilogue: biases, time decay, activations, hy/cy ----
  #pragma unroll
  for (int nt = 0; nt < 2; nt++) {
    int h = col0 + wn * 32 + nt * 16 + l15;
    float bd = bdec[h];
    float bi = bias4[h];
    float bfg = bias4[h + HID];
    float bc = bias4[h + 2 * HID];
    float bo = bias4[h + 3 * HID];
    #pragma unroll
    for (int r = 0; r < 8; r++) {
      int b = row0 + wm * 16 + r + 8 * half16;
      float tv = tvec[b];
      float T = (tv != 0.0f) ? (1.0f / tv) : 0.0f;
      float cxv = cx[(size_t)b * HID + h];
      float cst = fast_tanh(accd[nt][r] + bd);
      float cadj = cxv + (T - 1.0f) * cst;
      float ig = sigm(accg[0][nt][r] + bi);
      float fg = sigm(accg[1][nt][r] + bfg);
      float cg = fast_tanh(accg[2][nt][r] + bc);
      float og = sigm(accg[3][nt][r] + bo);
      float cyv = fg * cadj + ig * cg;
      float hyv = og * fast_tanh(cyv);
      hy[(size_t)b * HID + h] = hyv;
      cy[(size_t)b * HID + h] = cyv;
    }
  }
}

// ---------------- launch ----------------

extern "C" void kernel_launch(void* const* d_in, const int* in_sizes, int n_in,
                              void* d_out, int out_size, void* d_ws, size_t ws_size,
                              hipStream_t stream) {
  (void)in_sizes; (void)n_in; (void)out_size; (void)ws_size;

  const float* input = (const float*)d_in[0];
  const float* t     = (const float*)d_in[1];
  const float* hx    = (const float*)d_in[2];
  const float* cx    = (const float*)d_in[3];
  const float* wih_f = (const float*)d_in[4];
  const float* whh_f = (const float*)d_in[5];
  const float* bih   = (const float*)d_in[6];
  const float* bhh   = (const float*)d_in[7];
  const float* wdec  = (const float*)d_in[8];
  const float* bdec  = (const float*)d_in[9];

  float* hy = (float*)d_out;
  float* cy = hy + (size_t)BATCH * HID;

  // workspace layout: all bf16 operands (L2-resident working set ~34 MB)
  __bf16* wih  = (__bf16*)d_ws;                      // [4H, I]
  __bf16* whh  = wih + (size_t)4 * HID * IN_DIM;     // [4H, H]
  __bf16* wdt  = whh + (size_t)4 * HID * HID;        // [H, H] = W_decomp^T
  __bf16* inbf = wdt + (size_t)HID * HID;            // [B, I]
  __bf16* hxbf = inbf + (size_t)BATCH * IN_DIM;      // [B, H]
  __bf16* cxbf = hxbf + (size_t)BATCH * HID;         // [B, H]
  float* bias4 = (float*)(cxbf + (size_t)BATCH * HID); // [4H] = bias_ih + bias_hh

  {
    int n = 4 * HID * IN_DIM;
    cvt_bf16_kernel<<<n / 4 / 256, 256, 0, stream>>>(wih_f, wih, n);
  }
  {
    int n = 4 * HID * HID;
    cvt_bf16_kernel<<<n / 4 / 256, 256, 0, stream>>>(whh_f, whh, n);
  }
  {
    int n = BATCH * IN_DIM;
    cvt_bf16_kernel<<<n / 4 / 256, 256, 0, stream>>>(input, inbf, n);
  }
  {
    int n = BATCH * HID;
    cvt_bf16_kernel<<<n / 4 / 256, 256, 0, stream>>>(hx, hxbf, n);
    cvt_bf16_kernel<<<n / 4 / 256, 256, 0, stream>>>(cx, cxbf, n);
  }
  transpose_cvt_kernel<<<dim3(HID / 32, HID / 32), 256, 0, stream>>>(wdec, wdt);
  bias_sum_kernel<<<(4 * HID) / 256, 256, 0, stream>>>(bih, bhh, bias4, 4 * HID);

  tlstm_fused<<<dim3(HID / 64, BATCH / 64), 256, 0, stream>>>(
      inbf, hxbf, cxbf, t, cx, wih, whh, wdt, bias4, bdec, hy, cy);
}
